// AlignGroup_65618510348894
// MI455X (gfx1250) — matrix operand order repaired and checked by reading, not compiled, not run on hardware
//
#include <hip/hip_runtime.h>
#include <hip/hip_bf16.h>

typedef __attribute__((ext_vector_type(16))) _Float16 v16h;
typedef __attribute__((ext_vector_type(8)))  float    v8f;

#define U_CNT   200000
#define I_CNT   100000
#define G_CNT   4096
#define D_DIM   64
#define B_CNT   4096
#define MEMP    10
#define TEMP_INV 5.0f
#define CLW     0.1f

// ---- WMMA fragment index helpers (CDNA5 ISA 7.12.2, 16-bit layouts) ----
// A 16x32 (MxK): lanes 0-15 -> M=lane, elems 0..7 = K(h8..h8+7), elems 8..15 = K(16+h8..23+h8)
// B 32x16 (KxN): lanes 0-15 -> N=lane, elem e -> K = e + h8 + (e >= 8 ? 8 : 0)
// C/D f32 16x16: VGPR r, lane l: M = r + 8*(l>=16), N = l&15.

// load 8 contiguous f32 (32B-aligned) -> half of a v16h fragment, as 2x b128 loads
__device__ __forceinline__ void ld8h(const float* __restrict__ p, v16h& v, int base) {
    const float4 x = ((const float4*)p)[0];
    const float4 y = ((const float4*)p)[1];
    v[base + 0] = (_Float16)x.x; v[base + 1] = (_Float16)x.y;
    v[base + 2] = (_Float16)x.z; v[base + 3] = (_Float16)x.w;
    v[base + 4] = (_Float16)y.x; v[base + 5] = (_Float16)y.y;
    v[base + 6] = (_Float16)y.z; v[base + 7] = (_Float16)y.w;
}

// ---- GEMM 1: final_g = overlap_graph (GxG) @ group_table (GxD) ----
// One wave per 16-row strip; A fragment loaded once per k-step, reused over 4 N-tiles.
__global__ __launch_bounds__(32) void k_gemm_overlap(const float* __restrict__ A,
                                                     const float* __restrict__ Bm,
                                                     float* __restrict__ C) {
    const int lane = threadIdx.x & 31;
    const int hi   = lane >> 4;
    const int mn   = lane & 15;
    const int h8   = hi ? 8 : 0;
    const int h16  = hi ? 16 : 0;
    const int row0 = blockIdx.x * 16;
    const float* __restrict__ arow = A + (size_t)(row0 + mn) * G_CNT;
    v8f acc[4] = {};
    for (int k0 = 0; k0 < G_CNT; k0 += 32) {
        v16h af;
        ld8h(arow + k0 + h8,      af, 0);
        ld8h(arow + k0 + 16 + h8, af, 8);
        if (k0 + 128 < G_CNT) __builtin_prefetch(arow + k0 + 128, 0, 1);
#pragma unroll
        for (int c = 0; c < 4; ++c) {
            v16h bf;
#pragma unroll
            for (int e = 0; e < 16; ++e)
                bf[e] = (_Float16)Bm[(size_t)(k0 + e + h8 + (e >= 8 ? 8 : 0)) * D_DIM + c * 16 + mn];
            acc[c] = __builtin_amdgcn_wmma_f32_16x16x32_f16(false, af, false, bf,
                                                            (short)0, acc[c], false, false);
        }
    }
#pragma unroll
    for (int c = 0; c < 4; ++c)
#pragma unroll
        for (int r = 0; r < 8; ++r)
            C[(size_t)(row0 + r + hi * 8) * D_DIM + c * 16 + mn] = acc[c][r];
}

// ---- GEMM 2: msg = msgX (Gx128) @ W^T (64x128) + bias ; final_g += msg ----
__global__ __launch_bounds__(32) void k_gemm_msg(const float* __restrict__ X,
                                                 const float* __restrict__ W,
                                                 const float* __restrict__ bias,
                                                 float* __restrict__ msg,
                                                 float* __restrict__ final_g) {
    const int lane = threadIdx.x & 31;
    const int hi   = lane >> 4;
    const int mn   = lane & 15;
    const int h8   = hi ? 8 : 0;
    const int h16  = hi ? 16 : 0;
    const int row0 = blockIdx.x * 16;
    const float* __restrict__ xrow = X + (size_t)(row0 + mn) * 128;
    v8f acc[4] = {};
#pragma unroll
    for (int k0 = 0; k0 < 128; k0 += 32) {
        v16h af;
        ld8h(xrow + k0 + h8,      af, 0);
        ld8h(xrow + k0 + 16 + h8, af, 8);
#pragma unroll
        for (int c = 0; c < 4; ++c) {
            // B[k][n] = W[n][k]  (msg = X @ W^T): two 8-f32 runs along k (h8, 16+h8)
            const float* __restrict__ wrow = W + (size_t)(c * 16 + mn) * 128 + k0 + h8;
            v16h bf;
            ld8h(wrow,     bf, 0);
            ld8h(wrow + 16, bf, 8);
            acc[c] = __builtin_amdgcn_wmma_f32_16x16x32_f16(false, af, false, bf,
                                                            (short)0, acc[c], false, false);
        }
    }
#pragma unroll
    for (int c = 0; c < 4; ++c) {
        const float b = bias[c * 16 + mn];
#pragma unroll
        for (int r = 0; r < 8; ++r) {
            const size_t idx = (size_t)(row0 + r + hi * 8) * D_DIM + c * 16 + mn;
            const float v = acc[c][r] + b;
            msg[idx] = v;
            final_g[idx] += v;   // exactly one writer per element
        }
    }
}

// ---- GEMM 3 (fused): ttl[i] += sum_j exp( (v1 @ v2^T)[i][j] / temp ) ----
__global__ __launch_bounds__(32) void k_infonce(const float* __restrict__ V1,
                                                const float* __restrict__ V2,
                                                float* __restrict__ ttl) {
    const int lane = threadIdx.x & 31;
    const int hi   = lane >> 4;
    const int mn   = lane & 15;
    const int h8   = hi ? 8 : 0;
    const int h16  = hi ? 16 : 0;
    const int row0 = blockIdx.x * 16;
    const int ccb  = blockIdx.y * 8;   // 8 column tiles per wave -> A reuse
    const float* __restrict__ v1row = V1 + (size_t)(row0 + mn) * 64;
    v16h a0, a1;
    ld8h(v1row + h8,           a0, 0);
    ld8h(v1row + 16 + h8,      a0, 8);
    ld8h(v1row + 32 + h8,      a1, 0);
    ld8h(v1row + 48 + h8,      a1, 8);
    for (int t = 0; t < 8; ++t) {
        const int col0 = (ccb + t) * 16;
        const float* __restrict__ v2row = V2 + (size_t)(col0 + mn) * 64;
        v16h b0, b1;
        ld8h(v2row + h8,           b0, 0);
        ld8h(v2row + 16 + h8,      b0, 8);
        ld8h(v2row + 32 + h8,      b1, 0);
        ld8h(v2row + 48 + h8,      b1, 8);
        v8f acc = {};
        acc = __builtin_amdgcn_wmma_f32_16x16x32_f16(false, a0, false, b0,
                                                     (short)0, acc, false, false);
        acc = __builtin_amdgcn_wmma_f32_16x16x32_f16(false, a1, false, b1,
                                                     (short)0, acc, false, false);
#pragma unroll
        for (int r = 0; r < 8; ++r) {
            float v = expf(acc[r] * TEMP_INV);
            v += __shfl_xor(v, 1, 32);
            v += __shfl_xor(v, 2, 32);
            v += __shfl_xor(v, 4, 32);
            v += __shfl_xor(v, 8, 32);          // sum over the 16-lane N group
            if (mn == 0) atomicAdd(&ttl[row0 + r + hi * 8], v);
        }
    }
}

// ---------------- sparse / elementwise helpers ----------------
__global__ void k_fill(float* __restrict__ p, long n) {
    long i = (long)blockIdx.x * blockDim.x + threadIdx.x;
    if (i < n) p[i] = 0.0f;
}

// dst[rows[e]*ld + coff + d] += vals[e] * src[cols[e]*64 + d]
__global__ void k_scatter(const float* __restrict__ vals, const int* __restrict__ rows,
                          const int* __restrict__ cols, const float* __restrict__ src,
                          float* __restrict__ dst, int nnz, int ld, int coff) {
    long i = (long)blockIdx.x * blockDim.x + threadIdx.x;
    int e = (int)(i >> 6), d = (int)(i & 63);
    if (e < nnz) {
        float v = vals[e] * src[(size_t)cols[e] * 64 + d];
        atomicAdd(&dst[(size_t)rows[e] * ld + coff + d], v);
    }
}

// dst[i][d] += src[idx[i]+ioff][d]
__global__ void k_gather_add(float* __restrict__ dst, const float* __restrict__ src,
                             const int* __restrict__ idx, int n, int ioff) {
    long i = (long)blockIdx.x * blockDim.x + threadIdx.x;
    if (i < (long)n * 64) {
        int r = (int)(i >> 6), d = (int)(i & 63);
        dst[i] += src[(size_t)(idx[r] + ioff) * 64 + d];
    }
}

__global__ void k_centers(const float* __restrict__ mem, float* __restrict__ centers) {
    long i = (long)blockIdx.x * blockDim.x + threadIdx.x;
    if (i < (long)B_CNT * 64) {
        int b = (int)(i >> 6), d = (int)(i & 63);
        float mx = mem[(size_t)(b * MEMP) * 64 + d], mn = mx;
        for (int j = 1; j < MEMP; ++j) {
            float x = mem[(size_t)(b * MEMP + j) * 64 + d];
            mx = fmaxf(mx, x); mn = fminf(mn, x);
        }
        centers[i] = 0.5f * (mx + mn);
    }
}

__global__ __launch_bounds__(32) void k_normalize(const float* __restrict__ src,
                                                  float* __restrict__ dst) {
    int b = blockIdx.x, lane = threadIdx.x;
    float x0 = src[(size_t)b * 64 + lane], x1 = src[(size_t)b * 64 + 32 + lane];
    float ss = x0 * x0 + x1 * x1;
    for (int m = 16; m >= 1; m >>= 1) ss += __shfl_xor(ss, m, 32);
    float inv = rsqrtf(ss);
    dst[(size_t)b * 64 + lane] = x0 * inv;
    dst[(size_t)b * 64 + 32 + lane] = x1 * inv;
}

__global__ __launch_bounds__(32) void k_diag(const float* __restrict__ v1,
                                             const float* __restrict__ v2,
                                             float* __restrict__ diag) {
    int b = blockIdx.x, lane = threadIdx.x;
    float s = v1[(size_t)b * 64 + lane] * v2[(size_t)b * 64 + lane]
            + v1[(size_t)b * 64 + 32 + lane] * v2[(size_t)b * 64 + 32 + lane];
    for (int m = 16; m >= 1; m >>= 1) s += __shfl_xor(s, m, 32);
    if (lane == 0) diag[b] = s;
}

__global__ void k_cl(const float* __restrict__ ttl, const float* __restrict__ diag,
                     float* __restrict__ acc) {
    int b = blockIdx.x * blockDim.x + threadIdx.x;
    if (b < B_CNT) {
        // -log(exp(diag/T)/ttl) = log(ttl) - diag/T
        atomicAdd(&acc[1], logf(ttl[b]) - diag[b] * TEMP_INV);
    }
}

__global__ __launch_bounds__(32) void k_bpr(const float* __restrict__ g,
                                            const float* __restrict__ ip,
                                            const float* __restrict__ in_,
                                            float* __restrict__ pred_out,
                                            float* __restrict__ acc) {
    int b = blockIdx.x, lane = threadIdx.x;
    float g0 = g[(size_t)b * 64 + lane], g1 = g[(size_t)b * 64 + 32 + lane];
    float p = g0 * ip[(size_t)b * 64 + lane] + g1 * ip[(size_t)b * 64 + 32 + lane];
    float n = g0 * in_[(size_t)b * 64 + lane] + g1 * in_[(size_t)b * 64 + 32 + lane];
    for (int m = 16; m >= 1; m >>= 1) { p += __shfl_xor(p, m, 32); n += __shfl_xor(n, m, 32); }
    if (lane == 0) {
        pred_out[b] = p;
        float x = n - p;
        float sp = (x > 20.0f) ? x : log1pf(expf(x));   // softplus
        atomicAdd(&acc[0], sp);
    }
}

__global__ void k_final(const float* __restrict__ acc, float* __restrict__ out) {
    if (threadIdx.x == 0 && blockIdx.x == 0)
        out[0] = acc[0] / (float)B_CNT + CLW * (acc[1] / (float)B_CNT);
}

// ---------------------------------------------------------------------------
extern "C" void kernel_launch(void* const* d_in, const int* in_sizes, int n_in,
                              void* d_out, int out_size, void* d_ws, size_t ws_size,
                              hipStream_t stream) {
    const float* user_table  = (const float*)d_in[0];
    const float* item_table  = (const float*)d_in[1];
    const float* group_table = (const float*)d_in[2];
    const float* overlap     = (const float*)d_in[3];
    const float* agg_w       = (const float*)d_in[4];   // L x 64 x 128
    const float* agg_b       = (const float*)d_in[5];   // L x 64
    const float* uh_vals     = (const float*)d_in[6];
    const float* ih_vals     = (const float*)d_in[7];
    const float* fh_vals     = (const float*)d_in[8];
    const int*   uh_rows     = (const int*)d_in[9];
    const int*   uh_cols     = (const int*)d_in[10];
    const int*   ih_rows     = (const int*)d_in[11];
    const int*   ih_cols     = (const int*)d_in[12];
    const int*   fh_rows     = (const int*)d_in[13];
    const int*   fh_cols     = (const int*)d_in[14];
    const int*   grp_in      = (const int*)d_in[15];
    const int*   pos_in      = (const int*)d_in[16];
    const int*   neg_in      = (const int*)d_in[17];
    const int*   member_ids  = (const int*)d_in[18];
    const int nnz_uh = in_sizes[6], nnz_ih = in_sizes[7], nnz_fh = in_sizes[8];

    float* W = (float*)d_ws;
    size_t off = 0;
    float* emb     = W + off; off += (size_t)(U_CNT + I_CNT) * 64;   // 19.2M
    float* final_g = W + off; off += (size_t)G_CNT * 64;
    float* msgX    = W + off; off += (size_t)G_CNT * 128;
    float* msg     = W + off; off += (size_t)G_CNT * 64;
    float* mem_emb = W + off; off += (size_t)B_CNT * MEMP * 64;
    float* ipos    = W + off; off += (size_t)B_CNT * 64;
    float* ineg    = W + off; off += (size_t)B_CNT * 64;
    float* gemb    = W + off; off += (size_t)B_CNT * 64;
    float* centers = W + off; off += (size_t)B_CNT * 64;
    float* v1      = W + off; off += (size_t)B_CNT * 64;
    float* v2      = W + off; off += (size_t)B_CNT * 64;
    float* ttl     = W + off; off += B_CNT;
    float* diag    = W + off; off += B_CNT;
    float* acc     = W + off; off += 8;

    auto nb = [](long n, int t) { return (unsigned)((n + t - 1) / t); };
    float* pred_out = (float*)d_out + 1;

    // zero accumulators this call owns
    k_fill<<<nb((long)B_CNT * MEMP * 64, 256), 256, 0, stream>>>(mem_emb, (long)B_CNT * MEMP * 64);
    k_fill<<<nb((long)B_CNT * 64, 256), 256, 0, stream>>>(ipos, (long)B_CNT * 64);
    k_fill<<<nb((long)B_CNT * 64, 256), 256, 0, stream>>>(ineg, (long)B_CNT * 64);
    k_fill<<<nb((long)B_CNT * 64, 256), 256, 0, stream>>>(gemb, (long)B_CNT * 64);
    k_fill<<<nb(B_CNT, 256), 256, 0, stream>>>(ttl, B_CNT);
    k_fill<<<1, 8, 0, stream>>>(acc, 8);

    // gathered-accumulator init: final_ui starts at the raw tables
    k_gather_add<<<nb((long)B_CNT * MEMP * 64, 256), 256, 0, stream>>>(mem_emb, user_table, member_ids, B_CNT * MEMP, 0);
    k_gather_add<<<nb((long)B_CNT * 64, 256), 256, 0, stream>>>(ipos, item_table, pos_in, B_CNT, 0);
    k_gather_add<<<nb((long)B_CNT * 64, 256), 256, 0, stream>>>(ineg, item_table, neg_in, B_CNT, 0);

    // group_emb0 = overlap @ group_table  (WMMA), written into final_g
    k_gemm_overlap<<<G_CNT / 16, 32, 0, stream>>>(overlap, group_table, final_g);

    for (int l = 0; l < 2; ++l) {
        const float* usrc = l ? emb : user_table;
        const float* isrc = l ? (emb + (size_t)U_CNT * 64) : item_table;
        k_fill<<<nb((long)G_CNT * 128, 256), 256, 0, stream>>>(msgX, (long)G_CNT * 128);
        k_scatter<<<nb((long)nnz_uh * 64, 256), 256, 0, stream>>>(uh_vals, uh_rows, uh_cols, usrc, msgX, nnz_uh, 128, 0);
        k_scatter<<<nb((long)nnz_ih * 64, 256), 256, 0, stream>>>(ih_vals, ih_rows, ih_cols, isrc, msgX, nnz_ih, 128, 64);
        // msg = msgX @ W_l^T + b_l ; final_g += msg  (WMMA)
        k_gemm_msg<<<G_CNT / 16, 32, 0, stream>>>(msgX, agg_w + (size_t)l * 64 * 128, agg_b + (size_t)l * 64, msg, final_g);
        k_fill<<<nb((long)(U_CNT + I_CNT) * 64, 256), 256, 0, stream>>>(emb, (long)(U_CNT + I_CNT) * 64);
        k_scatter<<<nb((long)nnz_fh * 64, 256), 256, 0, stream>>>(fh_vals, fh_rows, fh_cols, msg, emb, nnz_fh, 64, 0);
        // final_ui accumulation, only at the rows we will ever read
        k_gather_add<<<nb((long)B_CNT * MEMP * 64, 256), 256, 0, stream>>>(mem_emb, emb, member_ids, B_CNT * MEMP, 0);
        k_gather_add<<<nb((long)B_CNT * 64, 256), 256, 0, stream>>>(ipos, emb, pos_in, B_CNT, U_CNT);
        k_gather_add<<<nb((long)B_CNT * 64, 256), 256, 0, stream>>>(ineg, emb, neg_in, B_CNT, U_CNT);
    }

    // g_emb = final_g[group_inputs]
    k_gather_add<<<nb((long)B_CNT * 64, 256), 256, 0, stream>>>(gemb, final_g, grp_in, B_CNT, 0);
    // centers = (segment_max + segment_min)/2 over MEMP members
    k_centers<<<nb((long)B_CNT * 64, 256), 256, 0, stream>>>(mem_emb, centers);
    // InfoNCE
    k_normalize<<<B_CNT, 32, 0, stream>>>(centers, v1);
    k_normalize<<<B_CNT, 32, 0, stream>>>(gemb, v2);
    k_diag<<<B_CNT, 32, 0, stream>>>(v1, v2, diag);
    k_infonce<<<dim3(B_CNT / 16, B_CNT / (16 * 8)), 32, 0, stream>>>(v1, v2, ttl);  // WMMA + fused exp-rowsum
    k_cl<<<nb(B_CNT, 256), 256, 0, stream>>>(ttl, diag, acc);
    // BPR + pos_pred output
    k_bpr<<<B_CNT, 32, 0, stream>>>(gemb, ipos, ineg, pred_out, acc);
    k_final<<<1, 1, 0, stream>>>(acc, (float*)d_out);
}
